// CGCNNDOSPredictor_7292854469250
// MI455X (gfx1250) — compile-verified, matrix operand order checked by reading, not run on hardware
//
#include <hip/hip_runtime.h>
#include <hip/hip_bf16.h>

// ---------------- problem constants (match reference) ----------------
#define N_NODES   100000
#define N_EDGES   1600000
#define NUM_GRAPHS 128
#define ATOM_IN   92
#define ATOM_KP   96          // padded K for embed GEMM
#define FEA       64
#define EDGE_DIM  41
#define N_CONV    5
#define BN_EPS    1e-5f

typedef __attribute__((ext_vector_type(16))) __bf16 v16bf;
typedef __attribute__((ext_vector_type(8)))  __bf16 v8bf;
typedef __attribute__((ext_vector_type(8)))  float  v8f;

__device__ __forceinline__ float sigmoidf_(float x) { return 1.0f / (1.0f + __expf(-x)); }
__device__ __forceinline__ float softplusf_(float x) { return x > 20.f ? x : log1pf(__expf(x)); }

// ---------------------------------------------------------------------
// WMMA GEMM: C[M,N] = A[M,KP](bf16) @ Bt[N,KP]^T(bf16) + bias[N]
// KP compile-time (multiple of 32), NT = N-tiles per wave (A reuse).
// One wave32 computes a 16 x (16*NT) strip of C. Fully unrolled K loop,
// 128-bit fragment loads. Fragment layout per CDNA5 ISA 7.12.2:
//   A 16x32: lanes 0-15 row=lane, elems0-7=K(k0..+7), elems8-15=K(k0+16..+23)
//            lanes16-31 row=lane-16, elems0-7=K(k0+8..), elems8-15=K(k0+24..)
//   B mirrored with col = lane&15 (Bt rows are B columns -> contiguous K).
//   C/D f32: VGPR v -> row = 8*(lane>>4)+v, col = lane&15.
// ---------------------------------------------------------------------
template <int KP, int NT>
__global__ void gemm_wmma_bf16(const __bf16* __restrict__ A, const __bf16* __restrict__ Bt,
                               const float* __restrict__ bias, float* __restrict__ C,
                               __bf16* __restrict__ Cbf, int M, int N) {
  int wave = (int)((blockIdx.x * (long)blockDim.x + threadIdx.x) >> 5);
  int lane = threadIdx.x & 31;
  int mt = M >> 4;
  int ngrp = N / (16 * NT);
  if (wave >= mt * ngrp) return;            // uniform whole-wave exit (EXEC all-1 for WMMA)
  int tm = wave / ngrp, tg = wave % ngrp;
  int hi  = lane >> 4;
  int l15 = lane & 15;

  const v8bf* __restrict__ pA = (const v8bf*)(A + (size_t)(tm * 16 + l15) * KP);
  v8f zero = {};
  v8f acc[NT];
#pragma unroll
  for (int j = 0; j < NT; ++j) acc[j] = zero;

#pragma unroll
  for (int k0 = 0; k0 < KP; k0 += 32) {
    int ia0 = (k0 + hi * 8) >> 3;           // 16-byte chunk index: elems 0..7
    int ia1 = (k0 + 16 + hi * 8) >> 3;      // elems 8..15
    v16bf a = __builtin_shufflevector(pA[ia0], pA[ia1],
                                      0, 1, 2, 3, 4, 5, 6, 7, 8, 9, 10, 11, 12, 13, 14, 15);
#pragma unroll
    for (int j = 0; j < NT; ++j) {
      const v8bf* __restrict__ pB =
          (const v8bf*)(Bt + (size_t)((tg * NT + j) * 16 + l15) * KP);
      v16bf b = __builtin_shufflevector(pB[ia0], pB[ia1],
                                        0, 1, 2, 3, 4, 5, 6, 7, 8, 9, 10, 11, 12, 13, 14, 15);
      acc[j] = __builtin_amdgcn_wmma_f32_16x16x32_bf16(false, a, false, b,
                                                       (short)0, acc[j], false, false);
    }
  }

#pragma unroll
  for (int j = 0; j < NT; ++j) {
    int col = (tg * NT + j) * 16 + l15;
    float bv = bias ? bias[col] : 0.f;
#pragma unroll
    for (int v = 0; v < 8; ++v) {
      size_t idx = (size_t)(tm * 16 + hi * 8 + v) * N + col;
      float r = acc[j][v] + bv;
      C[idx] = r;
      if (Cbf) Cbf[idx] = (__bf16)r;
    }
  }
}

// ---------------------------------------------------------------------
// f32 -> bf16 convert with K zero-padding: dst[M,KP] from src[M,K].
// ---------------------------------------------------------------------
__global__ void cvt_bf16_pad(const float* __restrict__ src, __bf16* __restrict__ dst,
                             int M, int K, int KP) {
  long i = blockIdx.x * (long)blockDim.x + threadIdx.x;
  if (i >= (long)M * KP) return;
  int r = (int)(i / KP), k = (int)(i % KP);
  dst[i] = (__bf16)(k < K ? src[(long)r * K + k] : 0.f);
}

// Transposed bf16 embed weights: Wt[c][k] = emb_W[k][c], k padded 92->96.
__global__ void prep_embWt(const float* __restrict__ W, __bf16* __restrict__ Wt) {
  int i = blockIdx.x * blockDim.x + threadIdx.x;       // 64*96
  if (i >= FEA * ATOM_KP) return;
  int c = i / ATOM_KP, k = i % ATOM_KP;
  Wt[i] = (__bf16)(k < ATOM_IN ? W[k * FEA + c] : 0.f);
}

// Combined+transposed per-layer node weights, bf16:
//   WcombT[l][c][k] (c in [0,256), k in [0,64)):
//     c<64   : Wf[k][c]        (dst part of lin_f)    bias = fb
//     c<128  : Wf[64+k][c-64]  (src part of lin_f)
//     c<192  : Ws[k][c-128]    (dst part of lin_s)    bias = sb
//     else   : Ws[64+k][c-192] (src part of lin_s)
__global__ void prep_combT(const float* __restrict__ fW, const float* __restrict__ fb,
                           const float* __restrict__ sW, const float* __restrict__ sb,
                           __bf16* __restrict__ Wt, float* __restrict__ bcomb) {
  int i = blockIdx.x * blockDim.x + threadIdx.x;       // 5*256*64
  if (i >= N_CONV * 256 * 64) return;
  int l = i / (256 * 64);
  int rem = i % (256 * 64);
  int c = rem / 64, k = rem % 64;
  const float* Wf = fW + l * 169 * 64;
  const float* Ws = sW + l * 169 * 64;
  float v;
  if (c < 64)        v = Wf[k * 64 + c];
  else if (c < 128)  v = Wf[(64 + k) * 64 + (c - 64)];
  else if (c < 192)  v = Ws[k * 64 + (c - 128)];
  else               v = Ws[(64 + k) * 64 + (c - 192)];
  Wt[i] = (__bf16)v;
  if (k == 0) {
    float bvv = 0.f;
    if (c < 64) bvv = fb[l * 64 + c];
    else if (c >= 128 && c < 192) bvv = sb[l * 64 + (c - 128)];
    bcomb[l * 256 + c] = bvv;
  }
}

__global__ void zero_f32(float* __restrict__ p, long n) {
  long i = blockIdx.x * (long)blockDim.x + threadIdx.x;
  if (i < n) p[i] = 0.f;
}

// ---------------------------------------------------------------------
// Edge message kernel. One wave per edge (grid-stride, edge id is
// wave-uniform so src/dst/edge_attr loads scalarize). Lane handles
// channels c=lane and c=lane+32. Edge-part weights (41x64 x2) in LDS.
//   gate_f(c) = P[dst][c]     + P[src][64+c]  + ea . WfE[:,c]  (bias in P)
//   gate_s(c) = P[dst][128+c] + P[src][192+c] + ea . WsE[:,c]
//   agg[dst][c] += sigmoid(gate_f) * softplus(gate_s)
// ---------------------------------------------------------------------
__global__ void edge_msg(const int* __restrict__ ei, const float* __restrict__ ea,
                         const float* __restrict__ P,
                         const float* __restrict__ WfE, const float* __restrict__ WsE,
                         float* __restrict__ agg) {
  __shared__ float swf[EDGE_DIM * 64];
  __shared__ float sws[EDGE_DIM * 64];
  for (int i = threadIdx.x; i < EDGE_DIM * 64; i += blockDim.x) {
    swf[i] = WfE[i];
    sws[i] = WsE[i];
  }
  __syncthreads();

  int lane = threadIdx.x & 31;
  int wave = (int)((blockIdx.x * (long)blockDim.x + threadIdx.x) >> 5);
  int nw   = (int)((gridDim.x * (long)blockDim.x) >> 5);

  for (int e = wave; e < N_EDGES; e += nw) {
    int s = ei[e];
    int d = ei[N_EDGES + e];
    long pd = (long)d * 256, ps = (long)s * 256;
    float gf0 = P[pd + lane]        + P[ps + 64 + lane];
    float gf1 = P[pd + 32 + lane]   + P[ps + 96 + lane];
    float gs0 = P[pd + 128 + lane]  + P[ps + 192 + lane];
    float gs1 = P[pd + 160 + lane]  + P[ps + 224 + lane];
    const float* eap = ea + (long)e * EDGE_DIM;
#pragma unroll 1
    for (int k = 0; k < EDGE_DIM; ++k) {
      float ev = eap[k];                     // wave-uniform -> scalar load
      gf0 = fmaf(ev, swf[k * 64 + lane],      gf0);
      gf1 = fmaf(ev, swf[k * 64 + lane + 32], gf1);
      gs0 = fmaf(ev, sws[k * 64 + lane],      gs0);
      gs1 = fmaf(ev, sws[k * 64 + lane + 32], gs1);
    }
    float m0 = sigmoidf_(gf0) * softplusf_(gs0);
    float m1 = sigmoidf_(gf1) * softplusf_(gs1);
    atomicAdd(&agg[(long)d * 64 + lane],      m0);
    atomicAdd(&agg[(long)d * 64 + lane + 32], m1);
  }
}

// ---------------------------------------------------------------------
// Batch-norm statistics: stats[c] = sum over nodes, stats[64+c] = sumsq.
// ---------------------------------------------------------------------
__global__ void bn_stats(const float* __restrict__ agg, float* __restrict__ stats) {
  int c = threadIdx.x & 63;
  int sub = threadIdx.x >> 6;               // blockDim.x = 256 -> 4 sub-rows
  float s = 0.f, ss = 0.f;
  for (int n = blockIdx.x * 4 + sub; n < N_NODES; n += gridDim.x * 4) {
    float v = agg[(long)n * 64 + c];
    s += v; ss += v * v;
  }
  __shared__ float red[512];
  red[threadIdx.x] = s;
  red[256 + threadIdx.x] = ss;
  __syncthreads();
  if (sub == 0) {
    for (int i = 1; i < 4; ++i) { s += red[i * 64 + c]; ss += red[256 + i * 64 + c]; }
    atomicAdd(&stats[c], s);
    atomicAdd(&stats[64 + c], ss);
  }
}

// BN + residual; also emits bf16 mirror of h for the next layer's GEMM.
__global__ void bn_apply(const float* __restrict__ agg, const float* __restrict__ stats,
                         const float* __restrict__ gamma, const float* __restrict__ beta,
                         float* __restrict__ h, __bf16* __restrict__ hbf) {
  long i = blockIdx.x * (long)blockDim.x + threadIdx.x;
  if (i >= (long)N_NODES * 64) return;
  int c = (int)(i & 63);
  float inv_n = 1.0f / (float)N_NODES;
  float mu  = stats[c] * inv_n;
  float var = stats[64 + c] * inv_n - mu * mu;
  float rs  = rsqrtf(var + BN_EPS);
  float nv  = (agg[i] - mu) * rs * gamma[c] + beta[c] + h[i];
  h[i] = nv;
  hbf[i] = (__bf16)nv;
}

// ---------------------------------------------------------------------
// Global mean pool (sums + counts via atomics; divide happens in head).
// ---------------------------------------------------------------------
__global__ void pool_sum(const float* __restrict__ h, const int* __restrict__ batch,
                         float* __restrict__ sums, float* __restrict__ cnt) {
  int lane = threadIdx.x & 31;
  int wave = (int)((blockIdx.x * (long)blockDim.x + threadIdx.x) >> 5);
  int nw   = (int)((gridDim.x * (long)blockDim.x) >> 5);
  for (int n = wave; n < N_NODES; n += nw) {
    int g = batch[n];
    atomicAdd(&sums[(long)g * 64 + lane],      h[(long)n * 64 + lane]);
    atomicAdd(&sums[(long)g * 64 + lane + 32], h[(long)n * 64 + lane + 32]);
    if (lane == 0) atomicAdd(&cnt[g], 1.0f);
  }
}

// ---------------------------------------------------------------------
// Head MLP: 64 -> softplus 256 -> softplus 256 -> 384. One block / graph.
// ---------------------------------------------------------------------
__global__ void head_mlp(const float* __restrict__ sums, const float* __restrict__ cnt,
                         const float* __restrict__ W1, const float* __restrict__ b1,
                         const float* __restrict__ W2, const float* __restrict__ b2,
                         const float* __restrict__ W3, const float* __restrict__ b3,
                         float* __restrict__ out) {
  int g = blockIdx.x;
  int t = threadIdx.x;                      // blockDim.x = 384
  __shared__ float pm[64];
  __shared__ float g1[256];
  __shared__ float g2[256];
  if (t < 64) {
    float c = cnt[g];
    if (c < 1.f) c = 1.f;
    pm[t] = sums[(long)g * 64 + t] / c;
  }
  __syncthreads();
  if (t < 256) {
    float a = b1[t];
    for (int k = 0; k < 64; ++k) a = fmaf(pm[k], W1[k * 256 + t], a);
    g1[t] = softplusf_(a);
  }
  __syncthreads();
  if (t < 256) {
    float a = b2[t];
    for (int k = 0; k < 256; ++k) a = fmaf(g1[k], W2[k * 256 + t], a);
    g2[t] = softplusf_(a);
  }
  __syncthreads();
  {
    float a = b3[t];
    for (int k = 0; k < 256; ++k) a = fmaf(g2[k], W3[k * 384 + t], a);
    out[(long)g * 384 + t] = a;
  }
}

// ---------------------------------------------------------------------
extern "C" void kernel_launch(void* const* d_in, const int* in_sizes, int n_in,
                              void* d_out, int out_size, void* d_ws, size_t ws_size,
                              hipStream_t stream) {
  (void)in_sizes; (void)n_in; (void)out_size; (void)ws_size;
  const float* x        = (const float*)d_in[0];
  const int*   ei       = (const int*)  d_in[1];
  const float* eattr    = (const float*)d_in[2];
  const int*   batch    = (const int*)  d_in[3];
  const float* emb_W    = (const float*)d_in[4];
  const float* emb_b    = (const float*)d_in[5];
  const float* lin_f_W  = (const float*)d_in[6];
  const float* lin_f_b  = (const float*)d_in[7];
  const float* lin_s_W  = (const float*)d_in[8];
  const float* lin_s_b  = (const float*)d_in[9];
  const float* bn_gamma = (const float*)d_in[10];
  const float* bn_beta  = (const float*)d_in[11];
  const float* hW1 = (const float*)d_in[12];
  const float* hb1 = (const float*)d_in[13];
  const float* hW2 = (const float*)d_in[14];
  const float* hb2 = (const float*)d_in[15];
  const float* hW3 = (const float*)d_in[16];
  const float* hb3 = (const float*)d_in[17];
  float* out = (float*)d_out;

  // ---- workspace carve (all chunk sizes are multiples of 256 bytes) ----
  char* ws = (char*)d_ws;
  float*  h      = (float*) ws; ws += (size_t)N_NODES * 64 * 4;      // 25.6 MB
  float*  P      = (float*) ws; ws += (size_t)N_NODES * 256 * 4;     // 102.4 MB
  float*  agg    = (float*) ws; ws += (size_t)N_NODES * 64 * 4;      // 25.6 MB
  float*  stats  = (float*) ws; ws += 128 * 4;                       // contiguous w/ agg
  __bf16* xbf    = (__bf16*)ws; ws += (size_t)N_NODES * ATOM_KP * 2; // 19.2 MB
  __bf16* hbf    = (__bf16*)ws; ws += (size_t)N_NODES * 64 * 2;      // 12.8 MB
  __bf16* embWt  = (__bf16*)ws; ws += (size_t)FEA * ATOM_KP * 2;
  __bf16* WcombT = (__bf16*)ws; ws += (size_t)N_CONV * 256 * 64 * 2;
  float*  bcomb  = (float*) ws; ws += (size_t)N_CONV * 256 * 4;
  float*  psums  = (float*) ws; ws += (size_t)NUM_GRAPHS * 64 * 4;
  float*  pcnt   = (float*) ws; ws += NUM_GRAPHS * 4;                // contiguous w/ psums

  auto zero = [&](float* p, long n) {
    int blocks = (int)((n + 255) / 256);
    zero_f32<<<blocks, 256, 0, stream>>>(p, n);
  };

  // 0) operand prep (deterministic, every call)
  {
    long nx = (long)N_NODES * ATOM_KP;
    cvt_bf16_pad<<<(int)((nx + 255) / 256), 256, 0, stream>>>(x, xbf, N_NODES, ATOM_IN, ATOM_KP);
    prep_embWt<<<(FEA * ATOM_KP + 255) / 256, 256, 0, stream>>>(emb_W, embWt);
    prep_combT<<<(N_CONV * 256 * 64 + 255) / 256, 256, 0, stream>>>(
        lin_f_W, lin_f_b, lin_s_W, lin_s_b, WcombT, bcomb);
  }

  // 1) embedding: h = x @ emb_W + emb_b  [100000,96]@[96,64] (WMMA, bf16 mirror -> hbf)
  {
    long waves = (long)(N_NODES / 16) * (FEA / 64);    // NT=4 covers all 4 col tiles
    int blocks = (int)((waves * 32 + 255) / 256);
    gemm_wmma_bf16<ATOM_KP, 4><<<blocks, 256, 0, stream>>>(xbf, embWt, emb_b, h, hbf,
                                                           N_NODES, FEA);
  }

  // 2) five CGConv layers
  for (int l = 0; l < N_CONV; ++l) {
    zero(agg, (long)N_NODES * 64 + 128);               // agg + stats (contiguous)
    // node precompute: P = h @ Wcomb[l] + bcomb[l]  [100000,64]@[64,256] (WMMA)
    {
      long waves = (long)(N_NODES / 16) * (256 / 64);  // NT=4 -> 4 waves per row strip
      int blocks = (int)((waves * 32 + 255) / 256);
      gemm_wmma_bf16<64, 4><<<blocks, 256, 0, stream>>>(
          hbf, WcombT + (size_t)l * 256 * 64, bcomb + (size_t)l * 256, P, (__bf16*)nullptr,
          N_NODES, 256);
    }
    // edge messages + scatter-add
    const float* WfE = lin_f_W + (size_t)l * 169 * 64 + 128 * 64;
    const float* WsE = lin_s_W + (size_t)l * 169 * 64 + 128 * 64;
    edge_msg<<<2048, 256, 0, stream>>>(ei, eattr, P, WfE, WsE, agg);
    // batch-norm + residual (also refresh hbf for next layer)
    bn_stats<<<512, 256, 0, stream>>>(agg, stats);
    bn_apply<<<(int)(((long)N_NODES * 64 + 255) / 256), 256, 0, stream>>>(
        agg, stats, bn_gamma + (size_t)l * 64, bn_beta + (size_t)l * 64, h, hbf);
  }

  // 3) global mean pool
  zero(psums, (long)NUM_GRAPHS * 64 + NUM_GRAPHS);     // psums + pcnt (contiguous)
  pool_sum<<<512, 256, 0, stream>>>(h, batch, psums, pcnt);

  // 4) head MLP -> d_out [128, 384] (== [128,3,128] flat)
  head_mlp<<<NUM_GRAPHS, 384, 0, stream>>>(psums, pcnt, hW1, hb1, hW2, hb2, hW3, hb3, out);
}
